// PrimsSolver_87299505258627
// MI455X (gfx1250) — matrix operand order repaired and checked by reading, not compiled, not run on hardware
//
#include <hip/hip_runtime.h>
#include <hip/hip_bf16.h>

#define NN      256
#define LATD    32
#define NSTEPS  255
#define SLOPE   0.01f

typedef __attribute__((ext_vector_type(16))) _Float16 v16h;
typedef __attribute__((ext_vector_type(8)))  _Float16 v8h;
typedef __attribute__((ext_vector_type(8)))  float    v8f;

__device__ __forceinline__ float leaky_f(float x) { return x >= 0.f ? x : SLOPE * x; }

// ---------------------------------------------------------------------------
// One-time init: pairwise edge weights, zero h / prev_tree, W_m2 -> f16 transposed
// ---------------------------------------------------------------------------
__global__ void init_kernel(const float* __restrict__ X, const float* __restrict__ Wm2,
                            float* __restrict__ ew, float* __restrict__ h,
                            float* __restrict__ pt, _Float16* __restrict__ wm2t) {
    int t = blockIdx.x * 256 + threadIdx.x;          // 0..65535
    int s = t >> 8, d = t & 255;
    float dx = X[2 * s] - X[2 * d];
    float dy = X[2 * s + 1] - X[2 * d + 1];
    ew[t] = sqrtf(dx * dx + dy * dy);
    if (t < NN * LATD) h[t] = 0.f;
    if (t < NN) pt[t] = 0.f;
    if (t < LATD * LATD) {                           // wm2t[n][k] = W_m2[k][n]
        int n = t >> 5, k = t & 31;
        wm2t[n * 32 + k] = (_Float16)Wm2[k * 32 + n];
    }
}

// ---------------------------------------------------------------------------
// Step 0 only: encoded = relu([pt,h]@W_enc+b); A = enc@Wm1[0:32]; B = enc@Wm1[32:64]
// ---------------------------------------------------------------------------
__global__ void encode_kernel(const float* __restrict__ pt, const float* __restrict__ h,
                              const float* __restrict__ Wenc, const float* __restrict__ benc,
                              const float* __restrict__ Wm1,
                              float* __restrict__ enc, float* __restrict__ Am, float* __restrict__ Bm) {
    int n = threadIdx.x;
    float acc[32];
    float ptn = pt[n];
#pragma unroll
    for (int j = 0; j < 32; ++j) acc[j] = benc[j] + ptn * Wenc[j];
#pragma unroll
    for (int k = 0; k < 32; ++k) {
        float hk = h[n * 32 + k];
        const float* w = Wenc + (1 + k) * 32;
#pragma unroll
        for (int j = 0; j < 32; ++j) acc[j] += hk * w[j];
    }
#pragma unroll
    for (int j = 0; j < 32; ++j) { acc[j] = fmaxf(acc[j], 0.f); enc[n * 32 + j] = acc[j]; }

    float a[32];
#pragma unroll
    for (int j = 0; j < 32; ++j) a[j] = 0.f;
#pragma unroll
    for (int k = 0; k < 32; ++k) {
        const float* w = Wm1 + k * 32;
#pragma unroll
        for (int j = 0; j < 32; ++j) a[j] += acc[k] * w[j];
    }
#pragma unroll
    for (int j = 0; j < 32; ++j) Am[n * 32 + j] = a[j];
#pragma unroll
    for (int j = 0; j < 32; ++j) a[j] = 0.f;
#pragma unroll
    for (int k = 0; k < 32; ++k) {
        const float* w = Wm1 + (32 + k) * 32;
#pragma unroll
        for (int j = 0; j < 32; ++j) a[j] += acc[k] * w[j];
    }
#pragma unroll
    for (int j = 0; j < 32; ++j) Bm[n * 32 + j] = a[j];
}

// ---------------------------------------------------------------------------
// Hot kernel: per dst d, aggr[d] = leaky(max_s( leaky(A[d]+B[s]+ew*wr) @ W_m2 ))
// 256 blocks (one dst), 128 threads (4 waves). Fragments built directly in
// registers; WMMA f32_16x16x32_f16; only the cross-wave max uses LDS.
// ---------------------------------------------------------------------------
__global__ void msg_kernel(const float* __restrict__ Am, const float* __restrict__ Bm,
                           const float* __restrict__ ew, const float* __restrict__ Wm1,
                           const _Float16* __restrict__ wm2t, float* __restrict__ aggr) {
    __shared__ float sRed[4 * 64];

    int d    = blockIdx.x;
    int tid  = threadIdx.x, wave = tid >> 5, lane = tid & 31;
    int mrow = lane & 15;
    int hi16 = (lane >> 4) & 1;
    int kbA  = hi16 ? 8 : 0;     // A-matrix: K 0-7/16-23 (lanes 0-15), 8-15/24-31 (lanes 16-31)
    int kbB  = hi16 ? 8 : 0;     // B-matrix: same K order as A (0-7/16-23, 8-15/24-31)

    // B fragments (W_m2^T as f16), once per block, straight from global
    v16h b0, b1;
    {
        const v8h* p0 = (const v8h*)(wm2t + mrow * 32 + kbB);
        const v8h* p1 = (const v8h*)(wm2t + (16 + mrow) * 32 + kbB);
        v8h q0 = p0[0], q1 = p0[2], q2 = p1[0], q3 = p1[2];
#pragma unroll
        for (int i = 0; i < 8; ++i) { b0[i] = q0[i]; b0[8 + i] = q1[i]; b1[i] = q2[i]; b1[8 + i] = q3[i]; }
    }
    // A[d] and w_row fragments in this lane's K order
    float adf[16], wrf[16];
    {
        const float* ad = Am + d * 32 + kbA;
        const float* wr = Wm1 + 64 * 32 + kbA;
#pragma unroll
        for (int i = 0; i < 8; ++i) {
            adf[i] = ad[i]; adf[8 + i] = ad[16 + i];
            wrf[i] = wr[i]; wrf[8 + i] = wr[16 + i];
        }
    }
    const float* ewrow = ew + d * 256;   // symmetric: ew[d][s] == ew[s][d]

    float mx0 = -3.4e38f, mx1 = -3.4e38f;
#pragma unroll
    for (int t4 = 0; t4 < 4; ++t4) {
        int tile = wave * 4 + t4;        // each wave owns 4 src tiles, uniform
        int s    = tile * 16 + mrow;     // this lane's source row
        float ewv = ewrow[s];
        const float* br = Bm + s * 32 + kbA;
        float4 f0 = *(const float4*)(br);
        float4 f1 = *(const float4*)(br + 4);
        float4 f2 = *(const float4*)(br + 16);
        float4 f3 = *(const float4*)(br + 20);
        float bv[16] = { f0.x, f0.y, f0.z, f0.w, f1.x, f1.y, f1.z, f1.w,
                         f2.x, f2.y, f2.z, f2.w, f3.x, f3.y, f3.z, f3.w };
        v16h a;
#pragma unroll
        for (int i = 0; i < 16; ++i) {
            float v = adf[i] + bv[i] + ewv * wrf[i];
            v = v >= 0.f ? v : SLOPE * v;
            a[i] = (_Float16)v;
        }
        v8f c0 = {}, c1 = {};
        c0 = __builtin_amdgcn_wmma_f32_16x16x32_f16(false, a, false, b0, (short)0, c0, false, false);
        c1 = __builtin_amdgcn_wmma_f32_16x16x32_f16(false, a, false, b1, (short)0, c1, false, false);
#pragma unroll
        for (int r = 0; r < 8; ++r) { mx0 = fmaxf(mx0, c0[r]); mx1 = fmaxf(mx1, c1[r]); }
    }

    int c = lane & 15;
    sRed[wave * 64 + hi16 * 32 + c]      = mx0;      // cols 0..15
    sRed[wave * 64 + hi16 * 32 + c + 16] = mx1;      // cols 16..31
    __syncthreads();
    if (tid < 32) {
        float m = -3.4e38f;
#pragma unroll
        for (int w = 0; w < 4; ++w) {
            m = fmaxf(m, sRed[w * 64 + tid]);
            m = fmaxf(m, sRed[w * 64 + 32 + tid]);
        }
        aggr[d * 32 + tid] = leaky_f(m);             // max(leaky(x)) == leaky(max(x))
    }
}

// ---------------------------------------------------------------------------
// Fused per-step tail: h = leaky([enc,aggr]@W_u); logits -> first-argmax -> pt;
// then (unless last step) next step's encoder reusing h_new in registers.
// ---------------------------------------------------------------------------
__global__ void update_encode_kernel(const float* __restrict__ enc, const float* __restrict__ aggr,
                                     const float* __restrict__ Wu, const float* __restrict__ Wmst,
                                     const float* __restrict__ bmst,
                                     const float* __restrict__ Wenc, const float* __restrict__ benc,
                                     const float* __restrict__ Wm1,
                                     float* __restrict__ h, float* __restrict__ pt,
                                     float* __restrict__ Am, float* __restrict__ Bm,
                                     float* __restrict__ encOut, int do_encode) {
    __shared__ float sLog[256];
    __shared__ int   sBi;
    int n = threadIdx.x;
    float ptn_old = pt[n];

    float e[32], g[32], hn[32];
#pragma unroll
    for (int k = 0; k < 32; ++k) { e[k] = enc[n * 32 + k]; g[k] = aggr[n * 32 + k]; }
#pragma unroll
    for (int j = 0; j < 32; ++j) hn[j] = 0.f;
#pragma unroll
    for (int k = 0; k < 32; ++k) {
        const float* w = Wu + k * 32;
#pragma unroll
        for (int j = 0; j < 32; ++j) hn[j] += e[k] * w[j];
    }
#pragma unroll
    for (int k = 0; k < 32; ++k) {
        const float* w = Wu + (32 + k) * 32;
#pragma unroll
        for (int j = 0; j < 32; ++j) hn[j] += g[k] * w[j];
    }
    float lg = bmst[0];
#pragma unroll
    for (int j = 0; j < 32; ++j) {
        hn[j] = leaky_f(hn[j]);
        h[n * 32 + j] = hn[j];
        lg += e[j] * Wmst[j] + hn[j] * Wmst[32 + j];   // sigmoid monotone -> argmax(logits)
    }
    sLog[n] = lg;
    __syncthreads();
    if (n == 0) {
        float best = sLog[0]; int bi = 0;
        for (int i = 1; i < 256; ++i) if (sLog[i] > best) { best = sLog[i]; bi = i; }
        sBi = bi;
        pt[bi] = 1.f;                                  // cumulative, for later kernels
    }
    __syncthreads();
    if (!do_encode) return;

    // ---- next step's encoder (h row already live in hn) ----
    float ptn = (n == sBi) ? 1.f : ptn_old;
    float acc[32];
#pragma unroll
    for (int j = 0; j < 32; ++j) acc[j] = benc[j] + ptn * Wenc[j];
#pragma unroll
    for (int k = 0; k < 32; ++k) {
        const float* w = Wenc + (1 + k) * 32;
#pragma unroll
        for (int j = 0; j < 32; ++j) acc[j] += hn[k] * w[j];
    }
#pragma unroll
    for (int j = 0; j < 32; ++j) { acc[j] = fmaxf(acc[j], 0.f); encOut[n * 32 + j] = acc[j]; }

    float a[32];
#pragma unroll
    for (int j = 0; j < 32; ++j) a[j] = 0.f;
#pragma unroll
    for (int k = 0; k < 32; ++k) {
        const float* w = Wm1 + k * 32;
#pragma unroll
        for (int j = 0; j < 32; ++j) a[j] += acc[k] * w[j];
    }
#pragma unroll
    for (int j = 0; j < 32; ++j) Am[n * 32 + j] = a[j];
#pragma unroll
    for (int j = 0; j < 32; ++j) a[j] = 0.f;
#pragma unroll
    for (int k = 0; k < 32; ++k) {
        const float* w = Wm1 + (32 + k) * 32;
#pragma unroll
        for (int j = 0; j < 32; ++j) a[j] += acc[k] * w[j];
    }
#pragma unroll
    for (int j = 0; j < 32; ++j) Bm[n * 32 + j] = a[j];
}

// ---------------------------------------------------------------------------
// Final decode precompute: P[s] = [enc,h][s]@Wp1[0:64] + b_p1 ; Q[d] = [enc,h][d]@Wp1[64:128]
// ---------------------------------------------------------------------------
__global__ void pq_kernel(const float* __restrict__ enc, const float* __restrict__ h,
                          const float* __restrict__ Wp1, const float* __restrict__ bp1,
                          float* __restrict__ P, float* __restrict__ Q) {
    int n = threadIdx.x;
    float e[32], hv[32], acc[32];
#pragma unroll
    for (int k = 0; k < 32; ++k) { e[k] = enc[n * 32 + k]; hv[k] = h[n * 32 + k]; }
#pragma unroll
    for (int j = 0; j < 32; ++j) acc[j] = bp1[j];
#pragma unroll
    for (int k = 0; k < 32; ++k) {
        const float* w0 = Wp1 + k * 32;
        const float* w1 = Wp1 + (32 + k) * 32;
#pragma unroll
        for (int j = 0; j < 32; ++j) acc[j] += e[k] * w0[j] + hv[k] * w1[j];
    }
#pragma unroll
    for (int j = 0; j < 32; ++j) P[n * 32 + j] = acc[j];
#pragma unroll
    for (int j = 0; j < 32; ++j) acc[j] = 0.f;
#pragma unroll
    for (int k = 0; k < 32; ++k) {
        const float* w0 = Wp1 + (64 + k) * 32;
        const float* w1 = Wp1 + (96 + k) * 32;
#pragma unroll
        for (int j = 0; j < 32; ++j) acc[j] += e[k] * w0[j] + hv[k] * w1[j];
    }
#pragma unroll
    for (int j = 0; j < 32; ++j) Q[n * 32 + j] = acc[j];
}

// out[s*256+d] = relu(P[s]+Q[d]) . Wp2 + b_p2
__global__ void pred_kernel(const float* __restrict__ P, const float* __restrict__ Q,
                            const float* __restrict__ Wp2, const float* __restrict__ bp2,
                            float* __restrict__ out) {
    __shared__ float sP[32], sW[32];
    int s = blockIdx.x, d = threadIdx.x;
    if (d < 32) { sP[d] = P[s * 32 + d]; sW[d] = Wp2[d]; }
    __syncthreads();
    float acc = bp2[0];
#pragma unroll
    for (int j = 0; j < 32; ++j) {
        float v = sP[j] + Q[d * 32 + j];
        acc += fmaxf(v, 0.f) * sW[j];
    }
    out[s * 256 + d] = acc;
}

// ---------------------------------------------------------------------------
extern "C" void kernel_launch(void* const* d_in, const int* in_sizes, int n_in,
                              void* d_out, int out_size, void* d_ws, size_t ws_size,
                              hipStream_t stream) {
    const float* X    = (const float*)d_in[0];
    const float* Wenc = (const float*)d_in[1];
    const float* benc = (const float*)d_in[2];
    const float* Wm1  = (const float*)d_in[3];
    const float* Wm2  = (const float*)d_in[4];
    const float* Wu   = (const float*)d_in[5];
    const float* Wmst = (const float*)d_in[6];
    const float* bmst = (const float*)d_in[7];
    const float* Wp1  = (const float*)d_in[8];
    const float* bp1  = (const float*)d_in[9];
    const float* Wp2  = (const float*)d_in[10];
    const float* bp2  = (const float*)d_in[11];

    float* ws   = (float*)d_ws;
    float* ew   = ws;                 // 65536
    float* enc  = ew   + 65536;       // 8192
    float* Am   = enc  + 8192;
    float* Bm   = Am   + 8192;
    float* aggr = Bm   + 8192;
    float* h    = aggr + 8192;
    float* P    = h    + 8192;
    float* Q    = P    + 8192;
    float* pt   = Q    + 8192;        // 256
    _Float16* wm2t = (_Float16*)(pt + 256);  // 1024 halfs

    init_kernel<<<256, 256, 0, stream>>>(X, Wm2, ew, h, pt, wm2t);
    encode_kernel<<<1, 256, 0, stream>>>(pt, h, Wenc, benc, Wm1, enc, Am, Bm);
    for (int it = 0; it < NSTEPS; ++it) {
        msg_kernel<<<256, 128, 0, stream>>>(Am, Bm, ew, Wm1, wm2t, aggr);
        update_encode_kernel<<<1, 256, 0, stream>>>(enc, aggr, Wu, Wmst, bmst,
                                                    Wenc, benc, Wm1,
                                                    h, pt, Am, Bm, enc,
                                                    (it < NSTEPS - 1) ? 1 : 0);
    }
    pq_kernel<<<1, 256, 0, stream>>>(enc, h, Wp1, bp1, P, Q);
    pred_kernel<<<256, 256, 0, stream>>>(P, Q, Wp2, bp2, (float*)d_out);
}
